// ALiBiMultiHeadAttention_61151744361017
// MI455X (gfx1250) — matrix operand order repaired and checked by reading, not compiled, not run on hardware
//
#include <hip/hip_runtime.h>

// ---------------------------------------------------------------------------
// ALiBi causal MHA forward for MI455X (gfx1250, wave32, WMMA bf16 16x16x32)
//   B=4, S=2048, E=1024, H=16, Dh=64.  Outputs: out [B,S,E] then attn [B,H,S,S]
// Pipeline:
//   k0: f32 -> bf16 convert of x and W_qkv               (elementwise)
//   k1: qkv = x @ W^T   (bf16 WMMA, f32 acc)
//       - 8 waves/block, 32x32 C-block per wave (4 WMMA / k-step)
//       - A panel (32x1024 bf16 = 64KB) async-staged to LDS
//         (global_load_async_to_lds_b128 + s_wait_asynccnt)
//       - epilogue also writes V^T for the AV GEMM
//   k2: per (b,h,16-row q-stripe): scores=QK^T/32+alibi (WMMA) -> LDS(128KB)
//       -> softmax in LDS -> write attn rows -> out = P @ V (WMMA via V^T)
// ---------------------------------------------------------------------------

typedef __attribute__((ext_vector_type(16))) __bf16        v16bf;
typedef __attribute__((ext_vector_type(8)))  float         v8f;
typedef __attribute__((ext_vector_type(4)))  unsigned int  v4u;
typedef __attribute__((ext_vector_type(4)))  float         v4f;
typedef __attribute__((address_space(3)))    char          lds_char_t;

#define S_LEN   2048
#define EMB     1024
#define H_CNT   16
#define THREE_E 3072
#define B_CNT   4
#define A_STRIDE 1032   // bf16 elems per LDS A row (1024 + 8 pad = 2064 B)

union ABu { v16bf v; v4u u[2]; unsigned short s[16]; };

static __device__ __forceinline__ unsigned short f2bf_bits(float f) {
  union { float f; unsigned u; } c; c.f = f;
  unsigned u = c.u + 0x7FFFu + ((c.u >> 16) & 1u);   // round-to-nearest-even
  return (unsigned short)(u >> 16);
}

static __device__ __forceinline__ v8f wmma_bf16(v16bf a, v16bf b, v8f c) {
  return __builtin_amdgcn_wmma_f32_16x16x32_bf16(false, a, false, b,
                                                 (short)0, c, false, false);
}

// ---------------- kernel 0: f32 -> bf16 ----------------
__global__ void __launch_bounds__(256)
cvt_f32_bf16(const float* __restrict__ in, unsigned short* __restrict__ outp, int n) {
  int i = blockIdx.x * 256 + threadIdx.x;
  if (i < n) outp[i] = f2bf_bits(in[i]);
}

// ---------------- kernel 1: QKV GEMM (M=8192, N=3072, K=1024) ----------------
__global__ void __launch_bounds__(256)
qkv_gemm(const unsigned short* __restrict__ xbf,
         const unsigned short* __restrict__ wbf,
         unsigned short* __restrict__ qkv,
         unsigned short* __restrict__ vt) {
  extern __shared__ unsigned short sA[];   // [32][A_STRIDE] bf16 (66048 B)
  const int tid  = threadIdx.x;
  const int lane = tid & 31;
  const int w    = tid >> 5;
  const int hi   = lane >> 4;
  const int l15  = lane & 15;
  const int m0   = blockIdx.x * 32;             // 32 rows per block
  const int f0   = blockIdx.y * 256 + w * 32;   // 32 feature cols per wave

  // ---- async-stage the contiguous 32x1024 bf16 A panel into LDS ----
  {
    const char* gbase = (const char*)(xbf + (size_t)m0 * EMB);
    const unsigned lds0 =
        (unsigned)(unsigned long long)(lds_char_t*)sA;   // LDS byte offset of sA
    #pragma unroll
    for (int it = 0; it < 16; ++it) {
      const int c   = it * 256 + tid;     // 16B chunk id, 4096 chunks total
      const int row = c >> 7;             // 128 chunks per 2048B source row
      const int col = c & 127;
      const unsigned long long ga =
          (unsigned long long)(gbase + (size_t)row * 2048 + col * 16);
      const unsigned loff = lds0 + (unsigned)(row * (A_STRIDE * 2) + col * 16);
      asm volatile("global_load_async_to_lds_b128 %0, %1, off"
                   :: "v"(loff), "v"(ga) : "memory");
    }
    asm volatile("s_wait_asynccnt 0x0" ::: "memory");
  }
  __syncthreads();

  const unsigned short* brow0 = wbf + (size_t)(f0 + l15) * EMB;
  const unsigned short* brow1 = brow0 + (size_t)16 * EMB;
  const unsigned short* ar0   = sA + (l15)      * A_STRIDE;   // A rows m0+l15
  const unsigned short* ar1   = sA + (16 + l15) * A_STRIDE;   // A rows m0+16+l15

  v8f c00 = {}, c01 = {}, c10 = {}, c11 = {};
  #pragma unroll 2
  for (int k0 = 0; k0 < EMB; k0 += 32) {
    ABu a0, a1, b0, b1;
    a0.u[0] = *(const v4u*)(ar0 + k0 + hi * 8);        // ds_load_b128
    a0.u[1] = *(const v4u*)(ar0 + k0 + 16 + hi * 8);
    a1.u[0] = *(const v4u*)(ar1 + k0 + hi * 8);
    a1.u[1] = *(const v4u*)(ar1 + k0 + 16 + hi * 8);
    b0.u[0] = *(const v4u*)(brow0 + k0 + hi * 8);      // global_load_b128
    b0.u[1] = *(const v4u*)(brow0 + k0 + 16 + hi * 8);
    b1.u[0] = *(const v4u*)(brow1 + k0 + hi * 8);
    b1.u[1] = *(const v4u*)(brow1 + k0 + 16 + hi * 8);
    c00 = wmma_bf16(a0.v, b0.v, c00);
    c01 = wmma_bf16(a0.v, b1.v, c01);
    c10 = wmma_bf16(a1.v, b0.v, c10);
    c11 = wmma_bf16(a1.v, b1.v, c11);
  }

  // ---- epilogue: qkv[m][f] row-major + transposed V copy vt[(b,h,d)][s] ----
  #pragma unroll
  for (int mt = 0; mt < 2; ++mt) {
    #pragma unroll
    for (int ft = 0; ft < 2; ++ft) {
      v8f c = mt ? (ft ? c11 : c10) : (ft ? c01 : c00);
      const int mbase = m0 + mt * 16;
      const int fbase = f0 + ft * 16;
      const int f     = fbase + l15;

      unsigned short bits[8];
      #pragma unroll
      for (int r = 0; r < 8; ++r) bits[r] = f2bf_bits(c[r]);

      #pragma unroll
      for (int r = 0; r < 8; ++r) {
        int mrow = mbase + r + hi * 8;   // C layout: lanes<16 -> M=r, else M=r+8
        qkv[(size_t)mrow * THREE_E + f] = bits[r];
      }

      if (fbase >= 2 * EMB) {            // V chunk -> also store transposed
        int d  = f - 2 * EMB;
        int hh = d >> 6;  d &= 63;
        int bi = mbase >> 11;            // mbase / S
        int s0 = (mbase & (S_LEN - 1)) + hi * 8;
        union { unsigned short us[8]; v4u u; } pk;
        #pragma unroll
        for (int r = 0; r < 8; ++r) pk.us[r] = bits[r];
        *(v4u*)(vt + (((size_t)(bi * H_CNT + hh) * 64 + d) * S_LEN + s0)) = pk.u;
      }
    }
  }
}

// ---------------- kernel 2: fused attention ----------------
__global__ void __launch_bounds__(256)
attn_fused(const unsigned short* __restrict__ qkv,
           const unsigned short* __restrict__ vt,
           float* __restrict__ out,
           float* __restrict__ attn) {
  extern __shared__ float sc[];        // [16][2048] f32 = 128 KB
  const int lane = threadIdx.x & 31;
  const int w    = threadIdx.x >> 5;
  const int hi   = lane >> 4;
  const int l15  = lane & 15;
  const int qt = blockIdx.x, h = blockIdx.y, b = blockIdx.z;
  const int q0 = qt * 16;
  const float slope  = exp2f(-0.5f * (float)(h + 1)); // alibi m[h] = 2^-((h+1)/2)
  const float NEGINF = -__builtin_inff();

  // ---- load Q A-fragments once (d = 0..63 -> two 16x32 frags) ----
  const unsigned short* qrow =
      qkv + ((size_t)(b * S_LEN + q0 + l15)) * THREE_E + EMB + h * 64;
  ABu aq0, aq1;
  aq0.u[0] = *(const v4u*)(qrow + hi * 8);
  aq0.u[1] = *(const v4u*)(qrow + 16 + hi * 8);
  aq1.u[0] = *(const v4u*)(qrow + 32 + hi * 8);
  aq1.u[1] = *(const v4u*)(qrow + 48 + hi * 8);

  // ---- phase 1: scores -> LDS.  wave w owns keys [256w, 256w+256) ----
  for (int t = 0; t < 16; ++t) {
    const int j0 = w * 256 + t * 16;
    const int j  = j0 + l15;
    if (j0 <= q0 + 15) {               // wave-uniform; skip fully-masked tiles
      const unsigned short* krow =
          qkv + ((size_t)(b * S_LEN + j0 + l15)) * THREE_E + h * 64;
      if (t < 15 && j0 + 16 <= q0 + 15)          // prefetch next K tile rows
        __builtin_prefetch(krow + 16 * THREE_E, 0, 1);
      ABu bk0, bk1;
      bk0.u[0] = *(const v4u*)(krow + hi * 8);
      bk0.u[1] = *(const v4u*)(krow + 16 + hi * 8);
      bk1.u[0] = *(const v4u*)(krow + 32 + hi * 8);
      bk1.u[1] = *(const v4u*)(krow + 48 + hi * 8);
      v8f c = {};
      c = wmma_bf16(aq0.v, bk0.v, c);
      c = wmma_bf16(aq1.v, bk1.v, c);
      #pragma unroll
      for (int r = 0; r < 8; ++r) {
        int m = r + hi * 8;
        int i = q0 + m;
        float s = (j <= i) ? (c[r] * 0.03125f + slope * (float)(j - i)) : NEGINF;
        sc[m * S_LEN + j] = s;
      }
    } else {
      #pragma unroll
      for (int r = 0; r < 8; ++r) sc[(r + hi * 8) * S_LEN + j] = NEGINF;
    }
  }
  __syncthreads();

  // ---- phase 2: row softmax in LDS + stream attn rows to HBM ----
  for (int mm = 0; mm < 2; ++mm) {
    const int m = w * 2 + mm;
    const int i = q0 + m;
    float* row = sc + m * S_LEN;
    float mx = NEGINF;
    for (int j = lane; j < S_LEN; j += 32) mx = fmaxf(mx, row[j]);
    #pragma unroll
    for (int off = 16; off > 0; off >>= 1) mx = fmaxf(mx, __shfl_xor(mx, off));
    float sum = 0.f;
    for (int j = lane; j < S_LEN; j += 32) {
      float e = __expf(row[j] - mx);
      row[j] = e; sum += e;
    }
    #pragma unroll
    for (int off = 16; off > 0; off >>= 1) sum += __shfl_xor(sum, off);
    const float inv = 1.f / sum;
    float* arow_g = attn + ((size_t)(b * H_CNT + h) * S_LEN + i) * S_LEN;
    for (int j = lane; j < S_LEN; j += 32) {
      float p = row[j] * inv;
      row[j]  = p;                     // keep normalized P for phase 3
      arow_g[j] = p;                   // coalesced f32 store (the 1.07GB output)
    }
  }
  __syncthreads();

  // ---- phase 3: out = P @ V via V^T (waves 0..3, one 16-wide d-tile each) ----
  if (w < 4) {
    const int d0 = w * 16;
    const unsigned short* vrow =
        vt + ((size_t)((b * H_CNT + h) * 64 + d0 + l15)) * S_LEN;
    v8f c = {};
    for (int jb = 0; jb < q0 + 16; jb += 32) {   // causal: stop at diagonal tile
      // A frag: P[m][jb..jb+31] from LDS, f32 -> bf16
      const float* pr = sc + (size_t)l15 * S_LEN + jb + hi * 8;
      v4f p0 = *(const v4f*)(pr);
      v4f p1 = *(const v4f*)(pr + 4);
      v4f p2 = *(const v4f*)(pr + 16);
      v4f p3 = *(const v4f*)(pr + 20);
      ABu a;
      #pragma unroll
      for (int e = 0; e < 4; ++e) {
        a.s[e]      = f2bf_bits(p0[e]);
        a.s[4 + e]  = f2bf_bits(p1[e]);
        a.s[8 + e]  = f2bf_bits(p2[e]);
        a.s[12 + e] = f2bf_bits(p3[e]);
      }
      // B frag: V^T row (fixed d, contiguous over j)
      ABu bv;
      bv.u[0] = *(const v4u*)(vrow + jb + hi * 8);
      bv.u[1] = *(const v4u*)(vrow + jb + 16 + hi * 8);
      c = wmma_bf16(a.v, bv.v, c);
    }
    float* obase = out + ((size_t)(b * S_LEN + q0)) * EMB + h * 64 + d0 + l15;
    #pragma unroll
    for (int r = 0; r < 8; ++r) {
      int m = r + hi * 8;
      obase[(size_t)m * EMB] = c[r];
    }
  }
}

// ---------------- launch ----------------
extern "C" void kernel_launch(void* const* d_in, const int* in_sizes, int n_in,
                              void* d_out, int out_size, void* d_ws, size_t ws_size,
                              hipStream_t stream) {
  (void)in_sizes; (void)n_in; (void)out_size; (void)ws_size;
  const float* x = (const float*)d_in[0];        // [4,2048,1024]
  const float* W = (const float*)d_in[1];        // [3072,1024]

  float* out  = (float*)d_out;                                   // [4,2048,1024]
  float* attn = out + (size_t)B_CNT * S_LEN * EMB;               // [4,16,2048,2048]

  // workspace carve-up (bf16 buffers): x 16MB | W 6MB | qkv 48MB | V^T 16MB
  unsigned short* xbf  = (unsigned short*)d_ws;
  unsigned short* wbf  = xbf  + (size_t)B_CNT * S_LEN * EMB;     // 8,388,608
  unsigned short* qkvb = wbf  + (size_t)THREE_E * EMB;           // 3,145,728
  unsigned short* vtb  = qkvb + (size_t)B_CNT * S_LEN * THREE_E; // 25,165,824

  const int nx = B_CNT * S_LEN * EMB;        // 8,388,608
  const int nw = THREE_E * EMB;              // 3,145,728
  cvt_f32_bf16<<<(nx + 255) / 256, 256, 0, stream>>>(x, xbf, nx);
  cvt_f32_bf16<<<(nw + 255) / 256, 256, 0, stream>>>(W, wbf, nw);

  qkv_gemm<<<dim3(256, 12), 256, 32 * A_STRIDE * 2, stream>>>(xbf, wbf, qkvb, vtb);

  attn_fused<<<dim3(S_LEN / 16, H_CNT, B_CNT), 256,
               16 * S_LEN * sizeof(float), stream>>>(qkvb, vtb, out, attn);
}
